// FasterRCNN_38414187495575
// MI455X (gfx1250) — matrix operand order repaired and checked by reading, not compiled, not run on hardware
//
#include <hip/hip_runtime.h>
#include <hip/hip_bf16.h>
#include <math.h>

typedef _Float16 v16h __attribute__((ext_vector_type(16)));
typedef _Float16 v8h  __attribute__((ext_vector_type(8)));
typedef float    v8f  __attribute__((ext_vector_type(8)));
typedef float    v4f  __attribute__((ext_vector_type(4)));

#define IMGF     512.0f
#define CCH      256
#define BATCH    2
#define PRE_NMS  1000
#define POST_NMS 256
#define NUM_A    65472
#define OUTP     7
#define NC1      81
#define IOU_T    0.7f

static __device__ __forceinline__ int imini(int a, int b) { return a < b ? a : b; }
static __device__ __forceinline__ int imaxi(int a, int b) { return a > b ? a : b; }

// ============================================================================
// Generic WMMA GEMM / implicit conv:
//   dst[M,N] = act( im2col(src) @ wgt[K,N] + bias [+ up2(up)] )
//   src: [NB,H,W,Cin] f32, M = NB*H*W, K = ksize*ksize*Cin (ksize in {1,3})
//   Requirements (true for every call site): M % 128 == 0, Cin % 32 == 0.
// Tile: BM=128 x BN=64, KT=32; 256 threads = 8 wave32 waves (4 M-groups x 2
// N-groups), each wave owns a 32x32 output patch = 2x2 v_wmma_f32_16x16x32_f16.
// Global->LDS staging is register double-buffered: the next K-step's tiles are
// loaded into VGPRs while the current step's WMMAs run; the weight tile two
// steps ahead is prefetched (global_prefetch_b8).
// ============================================================================
#define BM  128
#define BN  64
#define KT  32
#define LAS 40   // LDS A row stride (halves), padded vs 32 to spread banks
#define LBS 40   // LDS B row stride (halves)

__global__ __launch_bounds__(256) void gemm_conv_wmma(
    const float* __restrict__ src, int NB, int H, int W, int Cin, int ksize,
    const float* __restrict__ wgt, const float* __restrict__ bias,
    float* __restrict__ dst, int N, const float* __restrict__ up, int relu)
{
    __shared__ _Float16 lA[BM * LAS];
    __shared__ _Float16 lB[BN * LBS];

    const int M    = NB * H * W;
    const int K    = ksize * ksize * Cin;
    const int tid  = threadIdx.x;
    const int lane = tid & 31;
    const int wid  = tid >> 5;
    const int wm   = wid & 3;   // 0..3 : 32-row group
    const int wn   = wid >> 2;  // 0..1 : 32-col group
    const int mBlock = blockIdx.y * BM;
    const int nBlock = blockIdx.x * BN;

    v8f acc[2][2];
#pragma unroll
    for (int i = 0; i < 2; i++)
#pragma unroll
        for (int j = 0; j < 2; j++)
#pragma unroll
            for (int r = 0; r < 8; r++) acc[i][j][r] = 0.0f;

    // staging roles
    const int arow = tid >> 1;        // 0..127 (A tile row)
    const int akc  = (tid & 1) * 16;  // 0 or 16 (A k-chunk)
    const int brow = tid >> 3;        // 0..31  (B tile k)
    const int bnc  = (tid & 7) * 8;   // B n-chunk

    // decode output row for A staging (row -> b,y,x); once per thread
    const int gmA = mBlock + arow;
    int ab, ay, ax;
    {
        int r = (gmA < M) ? gmA : 0;
        ab = r / (H * W); r -= ab * H * W;
        ay = r / W; ax = r - ay * W;
    }
    const float* aBase = src + ((size_t)(ab * H + ay) * W + ax) * Cin;  // center pixel
    const bool   rowOk = (gmA < M);

    // WMMA fragment coordinates (ISA 7.12.2 layouts)
    const int kbA  = (lane < 16) ? 0 : 8;   // A frag: halves 0-7 = K kbA..+7, 8-15 = +16
    const int rowA = lane & 15;
    const int kbB  = (lane < 16) ? 0 : 8;   // B frag: halves 0-7 = K kbB..+7, 8-15 = +16
    const int colB = lane & 15;

    // ---- register staging buffers (double buffer) ----
    v4f   aReg[4];
    float bReg[8];

    // A tile loader for (tap, c0): aReg <- 16 f32 (or zeros)
    auto loadA = [&](int tap, int c0) {
        int dy = 0, dx = 0;
        if (ksize == 3) { int q = tap / 3; dy = q - 1; dx = tap - q * 3 - 1; }
        int sy = ay + dy, sx = ax + dx;
        bool ok = rowOk && (sy >= 0) && (sy < H) && (sx >= 0) && (sx < W);
        if (ok) {
            const float* p = aBase + ((size_t)dy * W + dx) * Cin + c0;
#pragma unroll
            for (int j = 0; j < 4; j++) aReg[j] = *(const v4f*)(p + 4 * j);
        } else {
#pragma unroll
            for (int j = 0; j < 4; j++)
#pragma unroll
                for (int e = 0; e < 4; e++) aReg[j][e] = 0.0f;
        }
    };
    // B tile loader for k-step base kb: bReg <- wgt[kb+brow][nBlock+bnc..+7]
    auto loadB = [&](int kb) {
        const float* p = wgt + (size_t)(kb + brow) * N + nBlock + bnc;
        if (nBlock + bnc + 8 <= N) {
            v4f v0 = *(const v4f*)(p);
            v4f v1 = *(const v4f*)(p + 4);
#pragma unroll
            for (int e = 0; e < 4; e++) { bReg[e] = v0[e]; bReg[4 + e] = v1[e]; }
        } else {
#pragma unroll
            for (int j = 0; j < 8; j++) {
                int n = nBlock + bnc + j;
                bReg[j] = (n < N) ? p[j] : 0.0f;
            }
        }
    };

    // thread's A k-position tracked incrementally: kk = tap*Cin + c0, kk = k0+akc
    int tapA = 0, c0A = akc;          // akc < 32 <= Cin
    loadA(tapA, c0A);
    loadB(0);

    for (int k0 = 0; k0 < K; k0 += KT) {
        // ---- commit staged registers to LDS (f32 -> f16) ----
        {
            _Float16* da = &lA[arow * LAS + akc];
#pragma unroll
            for (int j = 0; j < 4; j++) {
                da[4 * j + 0] = (_Float16)aReg[j][0];
                da[4 * j + 1] = (_Float16)aReg[j][1];
                da[4 * j + 2] = (_Float16)aReg[j][2];
                da[4 * j + 3] = (_Float16)aReg[j][3];
            }
#pragma unroll
            for (int j = 0; j < 8; j++)
                lB[(bnc + j) * LBS + brow] = (_Float16)bReg[j];
        }
        __syncthreads();

        // ---- issue next step's global loads (overlap with WMMA below) ----
        int kn = k0 + KT;
        if (kn < K) {
            c0A += KT;
            if (c0A >= Cin) { c0A -= Cin; tapA++; }
            loadA(tapA, c0A);
            loadB(kn);
            int kp = k0 + 2 * KT;     // prefetch weights two steps ahead
            if (kp < K)
                __builtin_prefetch(wgt + (size_t)(kp + brow) * N + nBlock + bnc, 0, 3);
        }

        // ---- WMMA compute: 2x2 tiles of 16x16x32 per wave ----
        v16h afrag[2], bfrag[2];
#pragma unroll
        for (int mt = 0; mt < 2; mt++) {
            int r = wm * 32 + mt * 16 + rowA;
            v8h lo = *(const v8h*)&lA[r * LAS + kbA];
            v8h hi = *(const v8h*)&lA[r * LAS + kbA + 16];
            v16h a;
#pragma unroll
            for (int j = 0; j < 8; j++) { a[j] = lo[j]; a[j + 8] = hi[j]; }
            afrag[mt] = a;
        }
#pragma unroll
        for (int nt = 0; nt < 2; nt++) {
            int c = wn * 32 + nt * 16 + colB;
            v8h lo = *(const v8h*)&lB[c * LBS + kbB];
            v8h hi = *(const v8h*)&lB[c * LBS + kbB + 16];
            v16h b;
#pragma unroll
            for (int j = 0; j < 8; j++) { b[j] = lo[j]; b[j + 8] = hi[j]; }
            bfrag[nt] = b;
        }
#pragma unroll
        for (int mt = 0; mt < 2; mt++)
#pragma unroll
            for (int nt = 0; nt < 2; nt++)
                acc[mt][nt] = __builtin_amdgcn_wmma_f32_16x16x32_f16(
                    false, afrag[mt], false, bfrag[nt], (short)0, acc[mt][nt],
                    false, false);
        __syncthreads();
    }

    // ---- epilogue: bias, optional up2-add, relu, store ----
#pragma unroll
    for (int mt = 0; mt < 2; mt++) {
#pragma unroll
        for (int nt = 0; nt < 2; nt++) {
            int col = nBlock + wn * 32 + nt * 16 + (lane & 15);
            if (col >= N) continue;
            float bv = bias ? bias[col] : 0.0f;
#pragma unroll
            for (int r = 0; r < 8; r++) {
                int row = mBlock + wm * 32 + mt * 16 + r + ((lane & 16) ? 8 : 0);
                if (row >= M) continue;
                float v = acc[mt][nt][r] + bv;
                if (up) {
                    int b  = row / (H * W);
                    int re = row - b * H * W;
                    int y  = re / W, x = re - (re / W) * W;
                    v += up[((size_t)(b * (H >> 1) + (y >> 1)) * (W >> 1) + (x >> 1)) * N + col];
                }
                if (relu) v = fmaxf(v, 0.0f);
                dst[(size_t)row * N + col] = v;
            }
        }
    }
}

// ============================================================================
// 2x2 max-pool (P5 -> c6)
// ============================================================================
__global__ void maxpool2_kernel(const float* __restrict__ in, float* __restrict__ out,
                                int B_, int Hi, int C)
{
    int i = blockIdx.x * blockDim.x + threadIdx.x;
    int Ho = Hi >> 1;
    int total = B_ * Ho * Ho * C;
    if (i >= total) return;
    int c = i % C; int t = i / C;
    int x = t % Ho; t /= Ho;
    int y = t % Ho; int b = t / Ho;
    const float* p = in + ((size_t)(b * Hi + 2 * y) * Hi + 2 * x) * C + c;
    float v = fmaxf(fmaxf(p[0], p[C]), fmaxf(p[(size_t)Hi * C], p[(size_t)Hi * C + C]));
    out[i] = v;
}

// ============================================================================
// RPN heads: obj (3) + dlt (12) 1x1 convs on relu'd rpn feature t.
// One block (32 thr) per spatial position; threads 0..14 each do one dot-256.
// ============================================================================
__global__ void rpn_heads_kernel(const float* __restrict__ t, int B_, int H, int W, int C,
                                 const float* __restrict__ objw, const float* __restrict__ objb,
                                 const float* __restrict__ dltw, const float* __restrict__ dltb,
                                 float* __restrict__ scores, float* __restrict__ deltas,
                                 int levelOff, int A_)
{
    int pos = blockIdx.x;         // 0..B*H*W-1
    int o   = threadIdx.x;
    if (o >= 15) return;
    int b   = pos / (H * W);
    int rem = pos - b * H * W;    // == y*W + x
    const float* tv = t + (size_t)pos * C;
    if (o < 3) {
        float s = objb[o];
        for (int c = 0; c < C; c++) s += tv[c] * objw[c * 3 + o];
        scores[(size_t)b * A_ + levelOff + rem * 3 + o] = s;
    } else {
        int q = o - 3;            // 0..11
        float s = dltb[q];
        for (int c = 0; c < C; c++) s += tv[c] * dltw[c * 12 + q];
        int anch = q >> 2, k = q & 3;
        deltas[((size_t)b * A_ + levelOff + rem * 3 + anch) * 4 + k] = s;
    }
}

// ============================================================================
// Box decode (anchors + deltas*VAR -> clipped yxyx boxes)
// ============================================================================
__global__ void decode_kernel(const float* __restrict__ anchors,
                              const float* __restrict__ deltas,
                              float* __restrict__ boxes, int B_, int A_)
{
    int i = blockIdx.x * blockDim.x + threadIdx.x;
    if (i >= B_ * A_) return;
    int a = i % A_;
    const float* an = anchors + (size_t)a * 4;
    const float* d  = deltas + (size_t)i * 4;
    float ah = an[2] - an[0], aw = an[3] - an[1];
    float acy = an[0] + 0.5f * ah, acx = an[1] + 0.5f * aw;
    float dy = d[0] * 0.1f, dx = d[1] * 0.1f, dh = d[2] * 0.2f, dw = d[3] * 0.2f;
    float cy = dy * ah + acy, cx = dx * aw + acx;
    float h = expf(dh) * ah, w = expf(dw) * aw;
    float* o = boxes + (size_t)i * 4;
    o[0] = fminf(fmaxf(cy - 0.5f * h, 0.0f), IMGF);
    o[1] = fminf(fmaxf(cx - 0.5f * w, 0.0f), IMGF);
    o[2] = fminf(fmaxf(cy + 0.5f * h, 0.0f), IMGF);
    o[3] = fminf(fmaxf(cx + 0.5f * w, 0.0f), IMGF);
}

// ============================================================================
// Top-1000 per batch: iterative argmax with LDS selected-bitmask (scores are
// never mutated -> no cross-CU global coherence hazard). Smallest-index
// tie-break matches jnp.argmax/top_k ordering.
// ============================================================================
__global__ __launch_bounds__(1024) void topk_kernel(
    const float* __restrict__ scores, const float* __restrict__ boxes,
    float* __restrict__ tb, float* __restrict__ ts)
{
    __shared__ unsigned mask[NUM_A / 32];   // 2046 words
    __shared__ float sv[1024];
    __shared__ int   si[1024];
    int b = blockIdx.x, tid = threadIdx.x;
    const float* s = scores + (size_t)b * NUM_A;
    for (int i = tid; i < NUM_A / 32; i += 1024) mask[i] = 0u;
    __syncthreads();
    for (int r = 0; r < PRE_NMS; r++) {
        float best = -__builtin_inff(); int bi = NUM_A;
        for (int i = tid; i < NUM_A; i += 1024) {
            if (mask[i >> 5] & (1u << (i & 31))) continue;
            float v = s[i];
            if (v > best) { best = v; bi = i; }   // ascending i: first max kept
        }
        sv[tid] = best; si[tid] = bi;
        __syncthreads();
        for (int o2 = 512; o2 > 0; o2 >>= 1) {
            if (tid < o2) {
                float v2 = sv[tid + o2]; int i2 = si[tid + o2];
                if (v2 > sv[tid] || (v2 == sv[tid] && i2 < si[tid])) { sv[tid] = v2; si[tid] = i2; }
            }
            __syncthreads();
        }
        if (tid == 0) {
            int j = si[0];
            ts[(size_t)b * PRE_NMS + r] = sv[0];
            const float* bj = boxes + ((size_t)b * NUM_A + j) * 4;
            float* o = tb + ((size_t)b * PRE_NMS + r) * 4;
            o[0] = bj[0]; o[1] = bj[1]; o[2] = bj[2]; o[3] = bj[3];
            mask[j >> 5] |= (1u << (j & 31));
        }
        __syncthreads();
    }
}

// ============================================================================
// Sequential NMS, 256 rounds, all state in LDS. One block per batch.
// ============================================================================
__global__ __launch_bounds__(1024) void nms_kernel(
    const float* __restrict__ tb, const float* __restrict__ ts,
    float* __restrict__ rois)
{
    __shared__ float by1[1024], bx1[1024], by2[1024], bx2[1024], ar[1024], sc[1024];
    __shared__ float rv[1024];
    __shared__ int   ri[1024];
    int b = blockIdx.x, tid = threadIdx.x;
    if (tid < PRE_NMS) {
        const float* p = tb + ((size_t)b * PRE_NMS + tid) * 4;
        by1[tid] = p[0]; bx1[tid] = p[1]; by2[tid] = p[2]; bx2[tid] = p[3];
        ar[tid]  = (p[2] - p[0]) * (p[3] - p[1]);
        sc[tid]  = ts[(size_t)b * PRE_NMS + tid];
    } else {
        by1[tid] = 0.f; bx1[tid] = 0.f; by2[tid] = 0.f; bx2[tid] = 0.f;
        ar[tid] = 0.f; sc[tid] = -__builtin_inff();
    }
    __syncthreads();
    for (int r = 0; r < POST_NMS; r++) {
        rv[tid] = sc[tid]; ri[tid] = tid;
        __syncthreads();
        for (int o2 = 512; o2 > 0; o2 >>= 1) {
            if (tid < o2) {
                if (rv[tid + o2] > rv[tid] ||
                    (rv[tid + o2] == rv[tid] && ri[tid + o2] < ri[tid])) {
                    rv[tid] = rv[tid + o2]; ri[tid] = ri[tid + o2];
                }
            }
            __syncthreads();
        }
        int j = ri[0];
        if (tid == 0) {
            float* o = rois + ((size_t)b * POST_NMS + r) * 4;
            o[0] = by1[j]; o[1] = bx1[j]; o[2] = by2[j]; o[3] = bx2[j];
        }
        float ty1 = by1[j], tx1 = bx1[j], ty2 = by2[j], tx2 = bx2[j], ta = ar[j];
        if (tid < PRE_NMS) {
            float yy1 = fmaxf(ty1, by1[tid]), xx1 = fmaxf(tx1, bx1[tid]);
            float yy2 = fminf(ty2, by2[tid]), xx2 = fminf(tx2, bx2[tid]);
            float inter = fmaxf(yy2 - yy1, 0.0f) * fmaxf(xx2 - xx1, 0.0f);
            float iou = inter / (ta + ar[tid] - inter + 1e-8f);
            if (iou > IOU_T || tid == j) sc[tid] = -__builtin_inff();
        }
        __syncthreads();
    }
}

// ============================================================================
// ROI align: one block per (b,roi), one channel per thread, 49 bins looped.
// ============================================================================
__global__ __launch_bounds__(256) void roialign_kernel(
    const float* __restrict__ rois,
    const float* __restrict__ f2, const float* __restrict__ f3,
    const float* __restrict__ f4, const float* __restrict__ f5,
    float* __restrict__ pooled)
{
    int br = blockIdx.x;                // b*POST + r
    int b  = br / POST_NMS;
    const float* rp = rois + (size_t)br * 4;
    float y1 = rp[0], x1 = rp[1], y2 = rp[2], x2 = rp[3];
    float area = (y2 - y1) * (x2 - x1);
    float lv   = floorf(4.0f + log2f(sqrtf(fmaxf(area, 0.0f)) / 224.0f + 1e-8f));
    int lvl    = (int)fminf(fmaxf(lv, 2.0f), 5.0f);
    const float* f; int H;
    if (lvl == 2)      { f = f2; H = 128; }
    else if (lvl == 3) { f = f3; H = 64; }
    else if (lvl == 4) { f = f4; H = 32; }
    else               { f = f5; H = 16; }
    float scl = 1.0f / (float)(1 << lvl);
    float ry1 = y1 * scl, rx1 = x1 * scl, ry2 = y2 * scl, rx2 = x2 * scl;
    int c = threadIdx.x;
    const float* base = f + (size_t)b * H * H * CCH;
    for (int by_ = 0; by_ < OUTP; by_++) {
        float tY = (by_ + 0.5f) / (float)OUTP;
        float yy = ry1 + tY * (ry2 - ry1) - 0.5f;
        float y0f = floorf(yy); float fy = yy - y0f;
        int y0  = imini(imaxi((int)y0f, 0), H - 1);
        int y1i = imini(imaxi((int)y0f + 1, 0), H - 1);
        for (int bx_ = 0; bx_ < OUTP; bx_++) {
            float tX = (bx_ + 0.5f) / (float)OUTP;
            float xx = rx1 + tX * (rx2 - rx1) - 0.5f;
            float x0f = floorf(xx); float fx = xx - x0f;
            int x0  = imini(imaxi((int)x0f, 0), H - 1);
            int x1i = imini(imaxi((int)x0f + 1, 0), H - 1);
            float v00 = base[((size_t)y0  * H + x0 ) * CCH + c];
            float v01 = base[((size_t)y0  * H + x1i) * CCH + c];
            float v10 = base[((size_t)y1i * H + x0 ) * CCH + c];
            float v11 = base[((size_t)y1i * H + x1i) * CCH + c];
            float v = v00 * (1 - fy) * (1 - fx) + v01 * (1 - fy) * fx
                    + v10 * fy * (1 - fx)       + v11 * fy * fx;
            pooled[(((size_t)br * OUTP + by_) * OUTP + bx_) * CCH + c] = v;
        }
    }
}

// ============================================================================
// Final: out[b,r,0:4]=box, out[b,r,4:85]=softmax(cls)
// ============================================================================
__global__ void finalize_kernel(const float* __restrict__ boxo,
                                const float* __restrict__ clso,
                                float* __restrict__ out, int R)
{
    int r = blockIdx.x * blockDim.x + threadIdx.x;
    if (r >= R) return;
    const float* bb = boxo + (size_t)r * 4;
    const float* cl = clso + (size_t)r * NC1;
    float* o = out + (size_t)r * (4 + NC1);
    for (int k = 0; k < 4; k++) o[k] = bb[k];
    float mx = -__builtin_inff();
    for (int k = 0; k < NC1; k++) mx = fmaxf(mx, cl[k]);
    float sum = 0.0f;
    for (int k = 0; k < NC1; k++) { float e = expf(cl[k] - mx); o[4 + k] = e; sum += e; }
    float inv = 1.0f / sum;
    for (int k = 0; k < NC1; k++) o[4 + k] *= inv;
}

// ============================================================================
extern "C" void kernel_launch(void* const* d_in, const int* in_sizes, int n_in,
                              void* d_out, int out_size, void* d_ws, size_t ws_size,
                              hipStream_t stream)
{
    (void)in_sizes; (void)n_in; (void)out_size; (void)ws_size;

    const float* P2 = (const float*)d_in[0];
    const float* P3 = (const float*)d_in[1];
    const float* P4 = (const float*)d_in[2];
    const float* P5 = (const float*)d_in[3];
    const float* anchors = (const float*)d_in[4];
    const float* lat_w[4] = {(const float*)d_in[5],  (const float*)d_in[7],
                             (const float*)d_in[9],  (const float*)d_in[11]};
    const float* lat_b[4] = {(const float*)d_in[6],  (const float*)d_in[8],
                             (const float*)d_in[10], (const float*)d_in[12]};
    const float* out_w[5] = {(const float*)d_in[13], (const float*)d_in[15],
                             (const float*)d_in[17], (const float*)d_in[19],
                             (const float*)d_in[21]};
    const float* out_b[5] = {(const float*)d_in[14], (const float*)d_in[16],
                             (const float*)d_in[18], (const float*)d_in[20],
                             (const float*)d_in[22]};
    const float* rpn_w = (const float*)d_in[23];
    const float* rpn_b = (const float*)d_in[24];
    const float* obj_w = (const float*)d_in[25];
    const float* obj_b = (const float*)d_in[26];
    const float* dlt_w = (const float*)d_in[27];
    const float* dlt_b = (const float*)d_in[28];
    const float* fc1_w = (const float*)d_in[29];
    const float* fc1_b = (const float*)d_in[30];
    const float* fc2_w = (const float*)d_in[31];
    const float* fc2_b = (const float*)d_in[32];
    const float* box_w = (const float*)d_in[33];
    const float* box_b = (const float*)d_in[34];
    const float* cls_w = (const float*)d_in[35];
    const float* cls_b = (const float*)d_in[36];
    float* out = (float*)d_out;

    // ---- workspace carve-out ----
    char* wsp = (char*)d_ws;
    size_t off = 0;
    auto alloc = [&](size_t bytes) -> float* {
        float* p = (float*)(wsp + off);
        off += (bytes + 255) & ~(size_t)255;
        return p;
    };
    float* l2 = alloc((size_t)BATCH * 128 * 128 * CCH * 4);
    float* l3 = alloc((size_t)BATCH * 64  * 64  * CCH * 4);
    float* l4 = alloc((size_t)BATCH * 32  * 32  * CCH * 4);
    float* l5 = alloc((size_t)BATCH * 16  * 16  * CCH * 4);
    float* c6 = alloc((size_t)BATCH * 8   * 8   * CCH * 4);
    float* f2 = alloc((size_t)BATCH * 128 * 128 * CCH * 4);
    float* f3 = alloc((size_t)BATCH * 64  * 64  * CCH * 4);
    float* f4 = alloc((size_t)BATCH * 32  * 32  * CCH * 4);
    float* f5 = alloc((size_t)BATCH * 16  * 16  * CCH * 4);
    float* f6 = alloc((size_t)BATCH * 8   * 8   * CCH * 4);
    float* scores = alloc((size_t)BATCH * NUM_A * 4);
    float* deltas = alloc((size_t)BATCH * NUM_A * 16);
    float* boxesd = alloc((size_t)BATCH * NUM_A * 16);
    float* tkb    = alloc((size_t)BATCH * PRE_NMS * 16);
    float* tks    = alloc((size_t)BATCH * PRE_NMS * 4);
    float* rois   = alloc((size_t)BATCH * POST_NMS * 16);
    float* pooled = alloc((size_t)BATCH * POST_NMS * OUTP * OUTP * CCH * 4);
    float* tbuf = l2;                         // rpn temp: l2 dead after f2
    float* fc1o = l3;                         // l3 dead after f3
    float* fc2o = l3 + (size_t)512 * 1024;
    float* boxo = l4;                         // l4 dead after f4
    float* clso = l4 + (size_t)512 * 8;

    const int R = BATCH * POST_NMS;           // 512

    // ---- FPN top-down laterals (fused up2-add) ----
    gemm_conv_wmma<<<dim3(4, (BATCH*16*16)/BM),   256, 0, stream>>>(P5, BATCH, 16, 16, CCH, 1, lat_w[3], lat_b[3], l5, CCH, nullptr, 0);
    gemm_conv_wmma<<<dim3(4, (BATCH*32*32)/BM),   256, 0, stream>>>(P4, BATCH, 32, 32, CCH, 1, lat_w[2], lat_b[2], l4, CCH, l5, 0);
    gemm_conv_wmma<<<dim3(4, (BATCH*64*64)/BM),   256, 0, stream>>>(P3, BATCH, 64, 64, CCH, 1, lat_w[1], lat_b[1], l3, CCH, l4, 0);
    gemm_conv_wmma<<<dim3(4, (BATCH*128*128)/BM), 256, 0, stream>>>(P2, BATCH, 128, 128, CCH, 1, lat_w[0], lat_b[0], l2, CCH, l3, 0);
    maxpool2_kernel<<<(BATCH*8*8*CCH + 255)/256, 256, 0, stream>>>(P5, c6, BATCH, 16, CCH);

    // ---- FPN 3x3 output convs (implicit GEMM, K=2304) ----
    gemm_conv_wmma<<<dim3(4, 256), 256, 0, stream>>>(l2, BATCH, 128, 128, CCH, 3, out_w[0], out_b[0], f2, CCH, nullptr, 0);
    gemm_conv_wmma<<<dim3(4, 64),  256, 0, stream>>>(l3, BATCH, 64,  64,  CCH, 3, out_w[1], out_b[1], f3, CCH, nullptr, 0);
    gemm_conv_wmma<<<dim3(4, 16),  256, 0, stream>>>(l4, BATCH, 32,  32,  CCH, 3, out_w[2], out_b[2], f4, CCH, nullptr, 0);
    gemm_conv_wmma<<<dim3(4, 4),   256, 0, stream>>>(l5, BATCH, 16,  16,  CCH, 3, out_w[3], out_b[3], f5, CCH, nullptr, 0);
    gemm_conv_wmma<<<dim3(4, 1),   256, 0, stream>>>(c6, BATCH, 8,   8,   CCH, 3, out_w[4], out_b[4], f6, CCH, nullptr, 0);

    // ---- RPN per level ----
    const float* feats[5] = {f2, f3, f4, f5, f6};
    const int    Hs[5]    = {128, 64, 32, 16, 8};
    const int    offs[5]  = {0, 49152, 61440, 64512, 65280};
    for (int L = 0; L < 5; L++) {
        int H = Hs[L];
        int M = BATCH * H * H;
        gemm_conv_wmma<<<dim3(4, M/BM), 256, 0, stream>>>(feats[L], BATCH, H, H, CCH, 3, rpn_w, rpn_b, tbuf, CCH, nullptr, 1);
        rpn_heads_kernel<<<M, 32, 0, stream>>>(tbuf, BATCH, H, H, CCH, obj_w, obj_b, dlt_w, dlt_b, scores, deltas, offs[L], NUM_A);
    }

    // ---- decode / top-k / NMS / ROI align ----
    decode_kernel<<<(BATCH*NUM_A + 255)/256, 256, 0, stream>>>(anchors, deltas, boxesd, BATCH, NUM_A);
    topk_kernel<<<BATCH, 1024, 0, stream>>>(scores, boxesd, tkb, tks);
    nms_kernel<<<BATCH, 1024, 0, stream>>>(tkb, tks, rois);
    roialign_kernel<<<R, 256, 0, stream>>>(rois, f2, f3, f4, f5, pooled);

    // ---- RCNN head: FC1(relu) -> FC2(relu) -> box / cls ----
    gemm_conv_wmma<<<dim3(16, R/BM), 256, 0, stream>>>(pooled, 1, R, 1, OUTP*OUTP*CCH, 1, fc1_w, fc1_b, fc1o, 1024, nullptr, 1);
    gemm_conv_wmma<<<dim3(16, R/BM), 256, 0, stream>>>(fc1o,   1, R, 1, 1024,          1, fc2_w, fc2_b, fc2o, 1024, nullptr, 1);
    gemm_conv_wmma<<<dim3(1,  R/BM), 256, 0, stream>>>(fc2o,   1, R, 1, 1024,          1, box_w, box_b, boxo, 4,    nullptr, 0);
    gemm_conv_wmma<<<dim3(2,  R/BM), 256, 0, stream>>>(fc2o,   1, R, 1, 1024,          1, cls_w, cls_b, clso, NC1,  nullptr, 0);

    finalize_kernel<<<(R + 255)/256, 256, 0, stream>>>(boxo, clso, out, R);
}
